// BitwiseNetwork_91336774517202
// MI455X (gfx1250) — matrix operand order repaired and checked by reading, not compiled, not run on hardware
//
#include <hip/hip_runtime.h>
#include <hip/hip_bf16.h>
#include <cstdint>
#include <cstddef>

// ---------------- problem constants ----------------
#define KW      1024      // conv kernel width
#define STR     256       // stride
#define CUT     513
#define TC      1026      // 2*CUT channels
#define NB      32        // batch
#define TIME    262144
#define NF      1029      // conv1 output frames
#define FP      1040      // frames padded to 65*16
#define TCP     1040      // channels padded to 65*16
#define XSTRIDE 267264    // padded input per batch (covers f<=1039 windows), mult of 8
#define HK      544       // padded reduction for mask GEMM (17*32)
#define CCP     1056      // padded channels for convT reduction (33*32)
#define OUTT    262144

typedef __bf16 bf16_t;
typedef __attribute__((ext_vector_type(16))) __bf16 v16bf;
typedef __attribute__((ext_vector_type(8)))  float  v8f;
typedef unsigned int u32x4 __attribute__((ext_vector_type(4)));
typedef int i32x4 __attribute__((ext_vector_type(4)));
typedef int i32x8 __attribute__((ext_vector_type(8)));

__device__ __forceinline__ v16bf frag2(const bf16_t* p0, const bf16_t* p1) {
  union { u32x4 u[2]; v16bf v; } x;
  x.u[0] = *reinterpret_cast<const u32x4*>(p0);
  x.u[1] = *reinterpret_cast<const u32x4*>(p1);
  return x.v;
}

__device__ __forceinline__ v8f wmma_bf16(v16bf a, v16bf b, v8f c) {
  return __builtin_amdgcn_wmma_f32_16x16x32_bf16(false, a, false, b, (short)0, c, false, false);
}

// ---------------- packing kernels ----------------
// x (B,1,TIME) f32 -> zero-padded bf16 [b][XSTRIDE], left pad 1024
__global__ void pack_x_kernel(const float* __restrict__ x, bf16_t* __restrict__ xp) {
  int i = blockIdx.x * 256 + threadIdx.x;            // 32*XSTRIDE threads
  int b = i / XSTRIDE, j = i % XSTRIDE;
  int src = j - 1024;
  float v = (src >= 0 && src < TIME) ? x[(size_t)b * TIME + src] : 0.0f;
  xp[i] = (bf16_t)v;
}

// conv1_w (TC,1,KW) f32 -> bf16 [TCP][KW], zero rows >= TC
__global__ void pack_w1_kernel(const float* __restrict__ w, bf16_t* __restrict__ wb) {
  int i = blockIdx.x * 256 + threadIdx.x;            // TCP*KW threads
  int k = i & (KW - 1); int c = i >> 10;
  float v = (c < TC) ? w[(size_t)c * KW + k] : 0.0f;
  wb[i] = (bf16_t)v;
}

// lin_w (CUT,CUT) f32 -> bf16 [528][HK] ; row i2 = B-column over k=j
__global__ void pack_lin_kernel(const float* __restrict__ w, bf16_t* __restrict__ lb) {
  int i = blockIdx.x * 256 + threadIdx.x;            // 528*HK threads
  int j = i % HK; int r = i / HK;
  float v = (r < CUT && j < CUT) ? w[(size_t)r * CUT + j] : 0.0f;
  lb[i] = (bf16_t)v;
}

// convT_w (TC,2,KW) -> bf16 B-matrix: wb[((o*256+r)*4+j)*CCP + c] = W[c,o,1024+r-256*(j+1)]
__global__ void pack_wt_kernel(const float* __restrict__ w, bf16_t* __restrict__ wb) {
  int i = blockIdx.x * 256 + threadIdx.x;            // 2*256*4*CCP threads
  int c = i % CCP; int t2 = i / CCP;
  int j = t2 & 3;  int t3 = t2 >> 2;
  int r = t3 & 255; int o = t3 >> 8;
  float v = 0.0f;
  if (c < TC) {
    int k = 1024 + r - 256 * (j + 1);                // always in [0,1024)
    v = w[((size_t)c * 2 + o) * KW + k];
  }
  wb[i] = (bf16_t)v;
}

__global__ void zero_u32_kernel(unsigned int* __restrict__ p) {
  p[blockIdx.x * 256 + threadIdx.x] = 0u;
}

// ---------------- conv1 as GEMM: T[c,f] = tanh(gamma[c] * sum_k W[c,k]*xwin[k,f]) ----------------
// A (weight 16x32 tile) is DMA'd to LDS by the Tensor Data Mover, double-buffered: tile k+1 is
// in flight while tile k feeds the WMMA. Fragments read with ds_load_b128.
// B (x windows) loaded straight from global (column f is k-contiguous).
__global__ void conv1_wmma_kernel(const bf16_t* __restrict__ xp, const bf16_t* __restrict__ w1,
                                  const float* __restrict__ in_gamma, bf16_t* __restrict__ tbf) {
  __shared__ __align__(64) bf16_t asmem[8][2][16 * 32];   // 2 x 1KB staging tiles per wave
  int wave = blockIdx.x * 8 + (threadIdx.x >> 5);
  if (wave >= NB * 65 * 65) return;
  int wid = threadIdx.x >> 5;
  int lane = threadIdx.x & 31;
  int ft = wave % 65; int t = wave / 65; int ct = t % 65; int b = t / 65;
  int c0 = ct * 16, f0 = ft * 16;
  int lm = lane & 15; bool hi = lane >= 16;

  const bf16_t* bCol = xp + (size_t)b * XSTRIDE + (size_t)(f0 + lm) * STR + (hi ? 8 : 0);

  // Tensor DMA descriptor, group1: data_size=2B, tensor 1024x1040 (stride 1024),
  // tile 32x16, no padding / no multicast / no iteration.
  const i32x8 g1 = { 0x00010000, 0x04000000, 0x04100000, 0x00200000, 16, 1024, 0, 0 };
  const i32x4 gz4 = { 0, 0, 0, 0 };
  const i32x8 gz8 = { 0, 0, 0, 0, 0, 0, 0, 0 };
  unsigned long long wbase =
      (unsigned long long)(uintptr_t)w1 + ((unsigned long long)c0 * KW) * 2ull;
  unsigned int lds0 = (unsigned int)(uintptr_t)(&asmem[wid][0][0]);
  unsigned int lds1 = (unsigned int)(uintptr_t)(&asmem[wid][1][0]);

  auto tdm_issue = [&](int kt) {
    unsigned long long ga = wbase + (unsigned long long)(kt * 32) * 2ull;
    u32x4 g0 = { 1u,                                   // count=1, user descriptor
                 (kt & 1) ? lds1 : lds0,               // LDS byte address (double buffer)
                 (unsigned int)ga,                     // global addr [31:0]
                 ((unsigned int)(ga >> 32) & 0x01FFFFFFu) | 0x80000000u }; // addr[56:32] | type=2
    __builtin_amdgcn_tensor_load_to_lds(g0, g1, gz4, gz4, gz8, 0);
  };

  tdm_issue(0);                                        // prime the pipeline
  v8f acc = {};
  for (int kt = 0; kt < 32; ++kt) {
    int k0 = kt * 32;
    if (kt < 31) {
      tdm_issue(kt + 1);                               // keep next tile in flight
      __builtin_amdgcn_s_wait_tensorcnt(1);            // in-order: tile kt has landed
    } else {
      __builtin_amdgcn_s_wait_tensorcnt(0);
    }
    asm volatile("" ::: "memory");                     // LDS rewritten by TDM: force re-load
    const bf16_t* aR = &asmem[wid][kt & 1][lm * 32 + (hi ? 8 : 0)];
    v16bf a = frag2(aR, aR + 16);                      // ds_load_b128 x2
    __builtin_prefetch(bCol + k0 + 64, 0, 0);
    v16bf bb = frag2(bCol + k0, bCol + k0 + 16);
    acc = wmma_bf16(a, bb, acc);
  }

  int f = f0 + lm;                                     // D: lane -> column N = f, VGPR -> row M = c
  if (f < NF) {
#pragma unroll
    for (int v = 0; v < 8; ++v) {
      int c = c0 + v + (hi ? 8 : 0);
      if (c < TC) {
        float val = tanhf(acc[v] * in_gamma[c]);
        tbf[((size_t)b * TC + c) * FP + f] = (bf16_t)val;
      }
    }
  }
}

// ---------------- pointwise 2->8->1 combo net -> h (bf16, GEMM-A layout [m=(b,f)][HK]) ----------------
__global__ void maskfeat_kernel(const bf16_t* __restrict__ tbf,
                                const float* __restrict__ c1w, const float* __restrict__ c1b,
                                const float* __restrict__ c2w, const float* __restrict__ c2b,
                                bf16_t* __restrict__ hbf) {
  int i = blockIdx.x * 256 + threadIdx.x;            // NB*FP*HK threads
  int ii = i % HK; int m = i / HK; int f = m % FP; int b = m / FP;
  float out = 0.0f;
  if (ii < CUT && f < NF) {
    float re = (float)tbf[((size_t)b * TC + ii) * FP + f];
    float im = (float)tbf[((size_t)b * TC + ii + CUT) * FP + f];
    float s = c2b[0];
#pragma unroll
    for (int o = 0; o < 8; ++o) {
      float h1 = fmaxf(c1w[o * 2] * re + c1w[o * 2 + 1] * im + c1b[o], 0.0f);
      s += c2w[o] * h1;
    }
    out = fmaxf(s, 0.0f);
  }
  hbf[(size_t)m * HK + ii] = (bf16_t)out;
}

// ---------------- mask GEMM (32928x513 @ 513x513) + fused mask-apply -> tm (convT A operand) ----------------
__global__ void maskgemm_wmma_kernel(const bf16_t* __restrict__ hbf, const bf16_t* __restrict__ linb,
                                     const float* __restrict__ lin_b, const float* __restrict__ fc_gamma,
                                     const bf16_t* __restrict__ tbf, bf16_t* __restrict__ tm) {
  int wave = blockIdx.x * 8 + (threadIdx.x >> 5);
  if (wave >= (NB * FP / 16) * 33) return;
  int lane = threadIdx.x & 31;
  int nt = wave % 33; int mt = wave / 33;
  int m0 = mt * 16, n0 = nt * 16;
  int lm = lane & 15; bool hi = lane >= 16;

  const bf16_t* aRow = hbf + (size_t)(m0 + lm) * HK + (hi ? 8 : 0);
  const bf16_t* bCol = linb + (size_t)(n0 + lm) * HK + (hi ? 8 : 0);

  v8f acc = {};
#pragma unroll 4
  for (int kt = 0; kt < 17; ++kt) {
    int k0 = kt * 32;
    __builtin_prefetch(aRow + k0 + 64, 0, 0);
    v16bf a = frag2(aRow + k0, aRow + k0 + 16);
    v16bf bb = frag2(bCol + k0, bCol + k0 + 16);
    acc = wmma_bf16(a, bb, acc);
  }

  int i2 = n0 + lm;                                  // column N = output feature
  int b = m0 / FP; int f0 = m0 % FP;                 // FP % 16 == 0: tile never crosses b
  if (i2 < CUT) {
    float bias = lin_b[i2], g = fc_gamma[i2];
#pragma unroll
    for (int v = 0; v < 8; ++v) {
      int f = f0 + v + (hi ? 8 : 0);
      if (f < NF) {
        float mk = 0.5f * (tanhf((acc[v] + bias) * g) + 1.0f);
        float tr = (float)tbf[((size_t)b * TC + i2) * FP + f];
        float ti = (float)tbf[((size_t)b * TC + i2 + CUT) * FP + f];
        size_t row = ((size_t)b * FP + f) * CCP;
        tm[row + i2]       = (bf16_t)(tr * mk);
        tm[row + i2 + CUT] = (bf16_t)(ti * mk);
      }
    }
  }
}

// ---------------- transposed conv as GEMM: M=16 q's, N=(2 outs x 8 residues), K=4 taps x CCP ----------
// N-blocked x2: each wave computes two residue-groups (16 residues) sharing the A fragment.
__global__ void convt_wmma_kernel(const bf16_t* __restrict__ tm, const bf16_t* __restrict__ wtb,
                                  float* __restrict__ out) {
  int wave = blockIdx.x * 8 + (threadIdx.x >> 5);
  if (wave >= NB * 64 * 16) return;
  int lane = threadIdx.x & 31;
  int rgp = wave & 15; int t = wave >> 4; int qt = t & 63; int b = t >> 6;
  int q0 = qt * 16, r0 = rgp * 16;
  int lm = lane & 15; bool hi = lane >= 16;
  int o = lm >> 3, rs = lm & 7;
  int rA = r0 + rs, rB = r0 + 8 + rs;                // columns of acc0 / acc1

  const bf16_t* aBase = tm + ((size_t)b * FP + q0 + lm + 1) * CCP;      // f = q + j + 1
  const bf16_t* bBase0 = wtb + (size_t)((o * 256 + rA) * 4) * CCP;
  const bf16_t* bBase1 = wtb + (size_t)((o * 256 + rB) * 4) * CCP;

  v8f acc0 = {}, acc1 = {};
#pragma unroll
  for (int j = 0; j < 4; ++j) {
    const bf16_t* aR = aBase + (size_t)j * CCP + (hi ? 8 : 0);
    const bf16_t* bC0 = bBase0 + (size_t)j * CCP + (hi ? 8 : 0);
    const bf16_t* bC1 = bBase1 + (size_t)j * CCP + (hi ? 8 : 0);
#pragma unroll 3
    for (int kt = 0; kt < 33; ++kt) {
      int k0 = kt * 32;
      __builtin_prefetch(aR + k0 + 64, 0, 0);
      v16bf a  = frag2(aR + k0, aR + k0 + 16);
      v16bf b0 = frag2(bC0 + k0, bC0 + k0 + 16);
      v16bf b1 = frag2(bC1 + k0, bC1 + k0 + 16);
      acc0 = wmma_bf16(a, b0, acc0);
      acc1 = wmma_bf16(a, b1, acc1);
    }
  }

#pragma unroll
  for (int v = 0; v < 8; ++v) {
    int q = q0 + v + (hi ? 8 : 0);
    size_t base = ((size_t)b * 2 + o) * OUTT + (size_t)q * 256;
    out[base + rA] = acc0[v];
    out[base + rB] = acc1[v];
  }
}

// ---------------- 2-way softmax over channel axis, in place ----------------
__global__ void softmax2_kernel(float* __restrict__ out) {
  int i = blockIdx.x * 256 + threadIdx.x;            // NB*OUTT threads
  int tau = i % OUTT; int b = i / OUTT;
  size_t i0 = ((size_t)b * 2) * OUTT + tau, i1 = i0 + OUTT;
  float y0 = out[i0], y1 = out[i1];
  float m = fmaxf(y0, y1);
  float e0 = __expf(y0 - m), e1 = __expf(y1 - m);
  float inv = 1.0f / (e0 + e1);
  out[i0] = e0 * inv;
  out[i1] = e1 * inv;
}

// ---------------- host ----------------
extern "C" void kernel_launch(void* const* d_in, const int* in_sizes, int n_in,
                              void* d_out, int out_size, void* d_ws, size_t ws_size,
                              hipStream_t stream) {
  const float* x       = (const float*)d_in[0];
  const float* conv1_w = (const float*)d_in[1];
  const float* in_gamma= (const float*)d_in[2];
  const float* comb1_w = (const float*)d_in[3];
  const float* comb1_b = (const float*)d_in[4];
  const float* comb2_w = (const float*)d_in[5];
  const float* comb2_b = (const float*)d_in[6];
  const float* lin_w   = (const float*)d_in[7];
  const float* lin_b   = (const float*)d_in[8];
  const float* fc_gamma= (const float*)d_in[9];
  const float* convT_w = (const float*)d_in[10];
  float* out = (float*)d_out;

  char* ws = (char*)d_ws;
  size_t off = 0;
  auto carve = [&](size_t bytes) { char* p = ws + off; off += (bytes + 255) & ~(size_t)255; return p; };

  bf16_t* xpad = (bf16_t*)carve((size_t)NB * XSTRIDE * 2);          // 17.1 MB
  bf16_t* w1b  = (bf16_t*)carve((size_t)TCP * KW * 2);              //  2.1 MB
  bf16_t* linb = (bf16_t*)carve((size_t)528 * HK * 2);              //  0.6 MB
  bf16_t* wtb  = (bf16_t*)carve((size_t)2 * 256 * 4 * CCP * 2);     //  4.3 MB
  bf16_t* tbf  = (bf16_t*)carve((size_t)NB * TC * FP * 2);          // 68.3 MB
  bf16_t* hbf  = (bf16_t*)carve((size_t)NB * FP * HK * 2);          // 36.2 MB
  bf16_t* tm   = (bf16_t*)carve((size_t)NB * FP * CCP * 2);         // 70.3 MB
  (void)ws_size;

  // pack / init
  pack_x_kernel  <<<(NB * XSTRIDE) / 256, 256, 0, stream>>>(x, xpad);
  pack_w1_kernel <<<(TCP * KW) / 256, 256, 0, stream>>>(conv1_w, w1b);
  pack_lin_kernel<<<(528 * HK) / 256, 256, 0, stream>>>(lin_w, linb);
  pack_wt_kernel <<<(2 * 256 * 4 * CCP) / 256, 256, 0, stream>>>(convT_w, wtb);
  zero_u32_kernel<<<((size_t)NB * FP * CCP * 2 / 4) / 256, 256, 0, stream>>>((unsigned int*)tm);

  // stage 1: strided conv as bf16 WMMA GEMM (double-buffered TDM-staged A) (+ gamma, tanh)
  conv1_wmma_kernel<<<(NB * 65 * 65) / 8, 256, 0, stream>>>(xpad, w1b, in_gamma, tbf);

  // stage 2: pointwise combo net -> GEMM operand
  maskfeat_kernel<<<(NB * FP * HK) / 256, 256, 0, stream>>>(tbf, comb1_w, comb1_b, comb2_w, comb2_b, hbf);

  // stage 3: 513x513 mask GEMM, fused bias/gamma/tanh/mask-apply -> convT operand
  maskgemm_wmma_kernel<<<((NB * FP / 16) * 33) / 8, 256, 0, stream>>>(hbf, linb, lin_b, fc_gamma, tbf, tm);

  // stage 4: transposed conv as WMMA GEMM (K = 4 taps x 1056 channels), N-blocked x2
  convt_wmma_kernel<<<(NB * 64 * 16) / 8, 256, 0, stream>>>(tm, wtb, out);

  // stage 5: softmax over the 2 output channels
  softmax2_kernel<<<(NB * OUTT) / 256, 256, 0, stream>>>(out);
}
